// DecoderCellRNN_52501680226811
// MI455X (gfx1250) — matrix operand order repaired and checked by reading, not compiled, not run on hardware
//
#include <hip/hip_runtime.h>

// ---------------------------------------------------------------------------
// DecoderCellRNN for MI455X (gfx1250): bf16 WMMA GEMMs, f32 accumulate.
// VOCAB=32000, EMBED=512, HIDDEN=1024, B=64, TGT=32, SRC=128
// GEMM inner loop is software-pipelined: next K-chunk's global_load_b128s are
// in flight while the current chunk's 4 v_wmma_f32_16x16x32_bf16 execute.
// ---------------------------------------------------------------------------

#define B_    64
#define TGT_  32
#define SRC_  128
#define HID_  1024
#define EMB_  512
#define VOC_  32000
#define XDIM_ (EMB_ + HID_)   // 1536
#define G3H_  (3 * HID_)      // 3072

typedef __attribute__((ext_vector_type(16))) __bf16 v16bf;
typedef __attribute__((ext_vector_type(8)))  float  v8f;
typedef __attribute__((ext_vector_type(4)))  unsigned int v4u;

union V16B { v16bf v; v4u u[2]; };

__device__ __forceinline__ unsigned short f2bf(float f) {
    unsigned int u = __float_as_uint(f);
    u += 0x7FFFu + ((u >> 16) & 1u);   // round-to-nearest-even
    return (unsigned short)(u >> 16);
}

__device__ __forceinline__ float sigm(float x) { return 1.0f / (1.0f + __expf(-x)); }

// --------------------------- prep kernels ----------------------------------

__global__ void f32_to_bf16_kernel(const float* __restrict__ src,
                                   unsigned short* __restrict__ dst, int n) {
    for (int i = blockIdx.x * blockDim.x + threadIdx.x; i < n;
         i += gridDim.x * blockDim.x)
        dst[i] = f2bf(src[i]);
}

__global__ void gather_emb_kernel(const int* __restrict__ tgt,
                                  const float* __restrict__ table,
                                  unsigned short* __restrict__ dst, int n) {
    for (int i = blockIdx.x * blockDim.x + threadIdx.x; i < n;
         i += gridDim.x * blockDim.x) {
        int tok = tgt[i / EMB_];
        dst[i] = f2bf(table[(size_t)tok * EMB_ + (i % EMB_)]);
    }
}

// ----------------------- attention + context (per step) --------------------

__global__ __launch_bounds__(128) void attn_ctx_kernel(
    const float* __restrict__ h,               // [B,H] f32
    const float* __restrict__ enc,             // [B,S,H] f32
    const unsigned short* __restrict__ emb_bf, // [B,T,E] bf16
    unsigned short* __restrict__ x_bf,         // [B, E+H] bf16
    int t)
{
    const int b = blockIdx.x, tid = threadIdx.x;
    __shared__ float hl[HID_];
    __shared__ float sc[SRC_];
    __shared__ float red[SRC_];

    const float* hb = h + (size_t)b * HID_;
    for (int k = tid; k < HID_; k += 128) hl[k] = hb[k];
    __syncthreads();

    const float* encb = enc + (size_t)b * SRC_ * HID_;
    // one score per thread (128 threads == SRC positions)
    {
        const float* e = encb + (size_t)tid * HID_;
        float s = 0.f;
        for (int k = 0; k < HID_; k += 4) {
            float4 ev = *reinterpret_cast<const float4*>(e + k);
            s += ev.x * hl[k] + ev.y * hl[k + 1] + ev.z * hl[k + 2] + ev.w * hl[k + 3];
        }
        sc[tid] = s; red[tid] = s;
    }
    __syncthreads();
    for (int off = 64; off > 0; off >>= 1) {
        if (tid < off) red[tid] = fmaxf(red[tid], red[tid + off]);
        __syncthreads();
    }
    const float mx = red[0];
    __syncthreads();
    float e = __expf(sc[tid] - mx);
    sc[tid] = e; red[tid] = e;
    __syncthreads();
    for (int off = 64; off > 0; off >>= 1) {
        if (tid < off) red[tid] += red[tid + off];
        __syncthreads();
    }
    const float inv = 1.0f / red[0];
    __syncthreads();

    // ctx: each thread owns 8 consecutive hidden dims
    const int k0 = tid * 8;
    float acc[8] = {0.f, 0.f, 0.f, 0.f, 0.f, 0.f, 0.f, 0.f};
    for (int s = 0; s < SRC_; ++s) {
        const float a = sc[s] * inv;
        const float* row = encb + (size_t)s * HID_ + k0;
        float4 r0 = *reinterpret_cast<const float4*>(row);
        float4 r1 = *reinterpret_cast<const float4*>(row + 4);
        acc[0] += a * r0.x; acc[1] += a * r0.y; acc[2] += a * r0.z; acc[3] += a * r0.w;
        acc[4] += a * r1.x; acc[5] += a * r1.y; acc[6] += a * r1.z; acc[7] += a * r1.w;
    }
    unsigned short* xb = x_bf + (size_t)b * XDIM_;
    for (int i = 0; i < 8; ++i) xb[EMB_ + k0 + i] = f2bf(acc[i]);

    // embedding half of x (already bf16)
    const unsigned short* eb = emb_bf + ((size_t)b * TGT_ + t) * EMB_;
    for (int i = tid * 4; i < tid * 4 + 4; ++i) xb[i] = eb[i];
}

// ----------------------------- WMMA GEMM -----------------------------------
// out[M,N] = A[M,K] @ W[N,K]^T (+bias). One wave computes a 16x64 strip
// (4 accumulator tiles, A registers reused across them). bf16 in, f32 out.
// Software-pipelined: K-chunk i+1 loads issue before chunk i's WMMAs.
// A layout (ISA 7.12.2, 16-bit A 16x32): lane half 0 -> K[0..7],K[16..23];
// half 1 -> K[8..15],K[24..31]. B: same K order as A, col = lane&15.

__device__ __forceinline__ void load_a(V16B& a, const unsigned short* aRow,
                                       int k0, int half) {
    const v4u* ap = reinterpret_cast<const v4u*>(aRow + k0 + half * 8);
    a.u[0] = ap[0];          // K chunk [k0 + half*8 .. +7]
    a.u[1] = ap[2];          // K chunk [k0 + 16 + half*8 .. +7]
}

__device__ __forceinline__ void load_b(V16B& bm, const unsigned short* wRow,
                                       size_t jOff, int k0, int half) {
    const v4u* bp = reinterpret_cast<const v4u*>(wRow + jOff + k0 + half * 8);
    bm.u[0] = bp[0];         // K [k0 + half*8 .. +7]
    bm.u[1] = bp[2];         // K [k0 + 16 + half*8 .. +7]
}

__global__ __launch_bounds__(128) void gemm_bf16_xt_kernel(
    const unsigned short* __restrict__ A,   // [M,K] bf16
    const unsigned short* __restrict__ W,   // [N,K] bf16
    float* __restrict__ out, int ldOut,
    const float* __restrict__ bias,
    int M, int N, int K)
{
    const int lane      = threadIdx.x & 31;
    const int waveInBlk = threadIdx.x >> 5;
    const int nQuads    = N >> 6;
    const int wid   = blockIdx.x * 4 + waveInBlk;
    const int mTile = wid / nQuads;
    const int nQuad = wid - mTile * nQuads;
    if (mTile * 16 >= M) return;

    const int r    = lane & 15;
    const int half = lane >> 4;

    const unsigned short* aRow = A + (size_t)(mTile * 16 + r) * K;
    const unsigned short* wRow = W + (size_t)(nQuad * 64 + r) * K;

    v8f acc[4] = {};
    V16B aCur, bCur[4];

    // prologue: chunk 0 in flight
    load_a(aCur, aRow, 0, half);
#pragma unroll
    for (int j = 0; j < 4; ++j) load_b(bCur[j], wRow, (size_t)j * 16 * K, 0, half);

    const int kIters = K >> 5;
    for (int it = 0; it < kIters - 1; ++it) {
        const int kn = (it + 1) << 5;
        V16B aNxt, bNxt[4];
        load_a(aNxt, aRow, kn, half);
#pragma unroll
        for (int j = 0; j < 4; ++j) load_b(bNxt[j], wRow, (size_t)j * 16 * K, kn, half);
        if (it + 2 < kIters) {   // pull chunk i+2 toward the WGP caches
            __builtin_prefetch(wRow + ((it + 2) << 5), 0, 3);
        }
#pragma unroll
        for (int j = 0; j < 4; ++j)
            acc[j] = __builtin_amdgcn_wmma_f32_16x16x32_bf16(
                false, aCur.v, false, bCur[j].v, (short)0, acc[j], false, false);
        aCur = aNxt;
#pragma unroll
        for (int j = 0; j < 4; ++j) bCur[j] = bNxt[j];
    }
    // epilogue chunk
#pragma unroll
    for (int j = 0; j < 4; ++j)
        acc[j] = __builtin_amdgcn_wmma_f32_16x16x32_bf16(
            false, aCur.v, false, bCur[j].v, (short)0, acc[j], false, false);

#pragma unroll
    for (int j = 0; j < 4; ++j) {
        const int nCol = nQuad * 64 + j * 16 + r;
        const float bv = bias ? bias[nCol] : 0.0f;
#pragma unroll
        for (int v = 0; v < 8; ++v) {
            const int mRow = mTile * 16 + v + half * 8;  // C/D layout
            out[(size_t)mRow * ldOut + nCol] = acc[j][v] + bv;
        }
    }
}

// --------------------------- GRU gate math ---------------------------------

__global__ void gru_gate_kernel(const float* __restrict__ gi,
                                const float* __restrict__ gh,
                                const float* __restrict__ b_ih,
                                const float* __restrict__ b_hh,
                                const float* __restrict__ hprev,
                                float* __restrict__ hnext,
                                unsigned short* __restrict__ h_bf)
{
    const int idx = blockIdx.x * blockDim.x + threadIdx.x;
    if (idx >= B_ * HID_) return;
    const int b = idx >> 10, k = idx & (HID_ - 1);
    const size_t base = (size_t)b * G3H_;
    const float ir = gi[base + k]            + b_ih[k];
    const float iz = gi[base + HID_ + k]     + b_ih[HID_ + k];
    const float in_= gi[base + 2 * HID_ + k] + b_ih[2 * HID_ + k];
    const float hr = gh[base + k]            + b_hh[k];
    const float hz = gh[base + HID_ + k]     + b_hh[HID_ + k];
    const float hn = gh[base + 2 * HID_ + k] + b_hh[2 * HID_ + k];
    const float rg = sigm(ir + hr);
    const float zg = sigm(iz + hz);
    const float ng = tanhf(in_ + rg * hn);
    const float h  = (1.0f - zg) * ng + zg * hprev[idx];
    hnext[idx] = h;
    h_bf[idx]  = f2bf(h);
}

__global__ void copy_f32_kernel(const float* __restrict__ src,
                                float* __restrict__ dst, int n) {
    for (int i = blockIdx.x * blockDim.x + threadIdx.x; i < n;
         i += gridDim.x * blockDim.x)
        dst[i] = src[i];
}

// ------------------------------- launch ------------------------------------

extern "C" void kernel_launch(void* const* d_in, const int* in_sizes, int n_in,
                              void* d_out, int out_size, void* d_ws, size_t ws_size,
                              hipStream_t stream) {
    (void)in_sizes; (void)n_in; (void)out_size; (void)ws_size;
    const int*   tgt    = (const int*)  d_in[0];
    const float* hidden = (const float*)d_in[1];
    const float* enc    = (const float*)d_in[2];
    const float* embT   = (const float*)d_in[3];
    const float* Wih    = (const float*)d_in[4];
    const float* Whh    = (const float*)d_in[5];
    const float* bih    = (const float*)d_in[6];
    const float* bhh    = (const float*)d_in[7];
    const float* Wout   = (const float*)d_in[8];
    const float* bout   = (const float*)d_in[9];
    float* out = (float*)d_out;

    // workspace carve-up (~86 MB)
    char* ws = (char*)d_ws;
    size_t o = 0;
    auto carve = [&](size_t bytes) -> char* {
        char* p = ws + o;
        o = (o + bytes + 255) & ~(size_t)255;
        return p;
    };
    unsigned short* WihB  = (unsigned short*)carve((size_t)G3H_ * XDIM_ * 2);
    unsigned short* WhhB  = (unsigned short*)carve((size_t)G3H_ * HID_ * 2);
    unsigned short* WoutB = (unsigned short*)carve((size_t)VOC_ * HID_ * 2);
    unsigned short* embB  = (unsigned short*)carve((size_t)B_ * TGT_ * EMB_ * 2);
    unsigned short* hBf   = (unsigned short*)carve((size_t)B_ * HID_ * 2);
    unsigned short* xBf   = (unsigned short*)carve((size_t)B_ * XDIM_ * 2);
    float* hbuf0 = (float*)carve((size_t)B_ * HID_ * 4);
    float* hbuf1 = (float*)carve((size_t)B_ * HID_ * 4);
    float* giBuf = (float*)carve((size_t)B_ * G3H_ * 4);
    float* ghBuf = (float*)carve((size_t)B_ * G3H_ * 4);
    float* hbufs[2] = {hbuf0, hbuf1};

    // one-time bf16 conversions (weights become L2-resident bf16 copies)
    f32_to_bf16_kernel<<<2048, 256, 0, stream>>>(Wih,  WihB,  G3H_ * XDIM_);
    f32_to_bf16_kernel<<<2048, 256, 0, stream>>>(Whh,  WhhB,  G3H_ * HID_);
    f32_to_bf16_kernel<<<4096, 256, 0, stream>>>(Wout, WoutB, VOC_ * HID_);
    f32_to_bf16_kernel<<<64,   256, 0, stream>>>(hidden, hBf, B_ * HID_);
    gather_emb_kernel<<<1024, 256, 0, stream>>>(tgt, embT, embB, B_ * TGT_ * EMB_);

    const int blkGates  = (4 * (G3H_ / 64)) / 4;   // 48
    const int blkLogits = (4 * (VOC_ / 64)) / 4;   // 500
    const int ldLogits  = TGT_ * VOC_;

    for (int t = 0; t < TGT_; ++t) {
        const float* hprev = (t == 0) ? hidden : hbufs[(t + 1) & 1];
        float*       hnext = hbufs[t & 1];

        attn_ctx_kernel<<<B_, 128, 0, stream>>>(hprev, enc, embB, xBf, t);

        gemm_bf16_xt_kernel<<<blkGates, 128, 0, stream>>>(
            xBf, WihB, giBuf, G3H_, nullptr, B_, G3H_, XDIM_);
        gemm_bf16_xt_kernel<<<blkGates, 128, 0, stream>>>(
            hBf, WhhB, ghBuf, G3H_, nullptr, B_, G3H_, HID_);

        gru_gate_kernel<<<(B_ * HID_ + 255) / 256, 256, 0, stream>>>(
            giBuf, ghBuf, bih, bhh, hprev, hnext, hBf);

        gemm_bf16_xt_kernel<<<blkLogits, 128, 0, stream>>>(
            hBf, WoutB, out + (size_t)t * VOC_, ldLogits, bout, B_, VOC_, HID_);
    }

    // final hidden state -> tail of d_out
    copy_f32_kernel<<<64, 256, 0, stream>>>(
        hbufs[(TGT_ - 1) & 1], out + (size_t)B_ * TGT_ * VOC_, B_ * HID_);
}
